// TinyTransformerLM_66357244723592
// MI455X (gfx1250) — matrix operand order repaired and checked by reading, not compiled, not run on hardware
//
#include <hip/hip_runtime.h>
#include <hip/hip_bf16.h>
#include <math.h>

// ---------------------------------------------------------------------------
// TinyTransformerLM forward for gfx1250 (MI455X).
// All dense algebra runs on v_wmma_f32_16x16x32_bf16 (wave32 WMMA, f32 acc).
// ---------------------------------------------------------------------------

typedef __bf16 bf16;
typedef bf16  v16bf __attribute__((ext_vector_type(16)));
typedef bf16  v8bf  __attribute__((ext_vector_type(8)));
typedef float v8f   __attribute__((ext_vector_type(8)));

union V16 { v16bf v; v8bf h[2]; };

// Model constants
constexpr int kV    = 32000;
constexpr int kD    = 1024;
constexpr int kH    = 16;
constexpr int kKVH  = 4;
constexpr int kL    = 4;
constexpr int kT    = 2048;
constexpr int kB    = 2;
constexpr int kHD   = kD / kH;          // 64
constexpr int kKVW  = kKVH * kHD;       // 256
constexpr int kQKVW = kD + 2 * kKVW;    // 1536
constexpr int kM    = kB * kT;          // 4096 token rows
constexpr float kEPS = 1e-5f;

// GEMM tiling
constexpr int BM = 128, BN = 128, BK = 32;
constexpr int LDSROW = 40;              // padded row stride (bf16 elems) -> conflict-free

// ---------------------------------------------------------------------------
// fp32 -> bf16 cast (grid-stride)
// ---------------------------------------------------------------------------
__global__ void cast_f32_bf16(const float* __restrict__ s, bf16* __restrict__ d, size_t n) {
  size_t i = (size_t)blockIdx.x * blockDim.x + threadIdx.x;
  size_t stride = (size_t)gridDim.x * blockDim.x;
  for (; i < n; i += stride) d[i] = (bf16)s[i];
}

// ---------------------------------------------------------------------------
// Embedding gather: x[row, :] = tok_emb[idx[row], :]
// ---------------------------------------------------------------------------
__global__ void embed_kernel(const float* __restrict__ tok, const int* __restrict__ idx,
                             float* __restrict__ x) {
  int row = blockIdx.x;
  const float* src = tok + (size_t)idx[row] * kD;
  float* dst = x + (size_t)row * kD;
  for (int c = threadIdx.x; c < kD; c += blockDim.x) dst[c] = src[c];
}

// ---------------------------------------------------------------------------
// LayerNorm over last dim (D=1024), writes bf16 (GEMM A operand)
// ---------------------------------------------------------------------------
__global__ __launch_bounds__(256)
void ln_bf16_kernel(const float* __restrict__ x, const float* __restrict__ s,
                    const float* __restrict__ b, bf16* __restrict__ out) {
  __shared__ float red[256];
  const int row = blockIdx.x, tid = threadIdx.x;
  const float* xr = x + (size_t)row * kD;

  float lsum = 0.f;
  for (int c = tid; c < kD; c += 256) lsum += xr[c];
  red[tid] = lsum; __syncthreads();
  for (int off = 128; off > 0; off >>= 1) {
    if (tid < off) red[tid] += red[tid + off];
    __syncthreads();
  }
  const float mu = red[0] * (1.0f / kD);
  __syncthreads();

  float lvar = 0.f;
  for (int c = tid; c < kD; c += 256) { float t = xr[c] - mu; lvar += t * t; }
  red[tid] = lvar; __syncthreads();
  for (int off = 128; off > 0; off >>= 1) {
    if (tid < off) red[tid] += red[tid + off];
    __syncthreads();
  }
  const float rstd = rsqrtf(red[0] * (1.0f / kD) + kEPS);

  bf16* o = out + (size_t)row * kD;
  for (int c = tid; c < kD; c += 256)
    o[c] = (bf16)((xr[c] - mu) * rstd * s[c] + b[c]);
}

// ---------------------------------------------------------------------------
// Tiled WMMA GEMM: C[M,N] = A[M,K] @ B + bias, optional GELU / residual.
// A, B in bf16; C fp32 and/or bf16. TRANSB: B is [N,K] row-major (tied head).
// Assumes M%128==0, N%128==0, K%32==0 (true for every call here).
// Fragment layouts follow CDNA5 ISA 7.12.2 exactly.
// ---------------------------------------------------------------------------
template <bool TRANSB, bool GELU, bool RES>
__global__ __launch_bounds__(256)
void gemm_wmma_kernel(const bf16* __restrict__ A, const bf16* __restrict__ Bm,
                      const float* __restrict__ bias, const float* __restrict__ res,
                      float* __restrict__ outF, bf16* __restrict__ outB,
                      int M, int N, int K) {
  __shared__ bf16 As[BM * LDSROW];
  __shared__ bf16 Bs[BN * LDSROW];   // stored K-transposed: Bs[n][k]

  const int t = threadIdx.x;
  const int mBase = blockIdx.y * BM;
  const int nBase = blockIdx.x * BN;
  const int waveId = t >> 5, lane = t & 31;
  const int wm = waveId >> 1;        // 0..3 -> 32-row band
  const int wn = waveId & 1;         // 0..1 -> 64-col band
  const int lm = lane & 15;
  const bool hiHalf = lane >= 16;

  v8f acc[2][4] = {};

  for (int k0 = 0; k0 < K; k0 += BK) {
    // ---- stage A tile (128x32): 512 x 8-elem vectors, 2 per thread ----
#pragma unroll
    for (int it = 0; it < 2; ++it) {
      int vid = t + it * 256;
      int r = vid >> 2, ck = (vid & 3) * 8;
      *(v8bf*)&As[r * LDSROW + ck] =
          *(const v8bf*)(A + (size_t)(mBase + r) * K + k0 + ck);
    }
    // ---- stage B tile into Bs[n][k] ----
    if (TRANSB) {
#pragma unroll
      for (int it = 0; it < 2; ++it) {
        int vid = t + it * 256;
        int r = vid >> 2, ck = (vid & 3) * 8;
        *(v8bf*)&Bs[r * LDSROW + ck] =
            *(const v8bf*)(Bm + (size_t)(nBase + r) * K + k0 + ck);
      }
    } else {
#pragma unroll
      for (int it = 0; it < 2; ++it) {
        int vid = t + it * 256;
        int kk = vid >> 4, cn = (vid & 15) * 8;
        v8bf tmp = *(const v8bf*)(Bm + (size_t)(k0 + kk) * N + nBase + cn);
#pragma unroll
        for (int i = 0; i < 8; ++i) Bs[(cn + i) * LDSROW + kk] = tmp[i];
      }
    }
    // prefetch next K-tile while this one computes (global_prefetch_b8)
    if (k0 + BK < K) {
      __builtin_prefetch(A + (size_t)(mBase + (t >> 1)) * K + k0 + BK, 0, 1);
      if (TRANSB)
        __builtin_prefetch(Bm + (size_t)(nBase + (t >> 1)) * K + k0 + BK, 0, 1);
      else
        __builtin_prefetch(Bm + (size_t)(k0 + BK + (t >> 4)) * N + nBase + (t & 15) * 8, 0, 1);
    }
    __syncthreads();

    // ---- gather fragments per ISA layout ----
    V16 aF[2], bF[4];
    const int kbA = hiHalf ? 8 : 0;
#pragma unroll
    for (int f = 0; f < 2; ++f) {
      const int m = wm * 32 + f * 16 + lm;
      aF[f].h[0] = *(const v8bf*)&As[m * LDSROW + kbA];        // K kbA..kbA+7
      aF[f].h[1] = *(const v8bf*)&As[m * LDSROW + kbA + 16];   // K kbA+16..kbA+23
    }
    const int kbB = hiHalf ? 8 : 0;
#pragma unroll
    for (int g = 0; g < 4; ++g) {
      const int n = wn * 64 + g * 16 + lm;
      bF[g].h[0] = *(const v8bf*)&Bs[n * LDSROW + kbB];
      bF[g].h[1] = *(const v8bf*)&Bs[n * LDSROW + kbB + 16];
    }
#pragma unroll
    for (int f = 0; f < 2; ++f)
#pragma unroll
      for (int g = 0; g < 4; ++g)
        acc[f][g] = __builtin_amdgcn_wmma_f32_16x16x32_bf16(
            false, aF[f].v, false, bF[g].v, (short)0, acc[f][g], false, false);
    __syncthreads();
  }

  // ---- epilogue: bias -> gelu -> residual -> store (C layout: ISA 7.12.2) ----
  const int rowOff = hiHalf ? 8 : 0;
#pragma unroll
  for (int f = 0; f < 2; ++f)
#pragma unroll
    for (int g = 0; g < 4; ++g) {
      const int col = nBase + wn * 64 + g * 16 + lm;
      const float bv = bias ? bias[col] : 0.f;
#pragma unroll
      for (int r = 0; r < 8; ++r) {
        const int row = mBase + wm * 32 + f * 16 + rowOff + r;
        const size_t o = (size_t)row * N + col;
        float v = acc[f][g][r] + bv;
        if (GELU) v = 0.5f * v * (1.0f + erff(v * 0.70710678118654752f));
        if (RES)  v += res[o];
        if (outF) outF[o] = v;
        if (outB) outB[o] = (bf16)v;
      }
    }
}

// ---------------------------------------------------------------------------
// Causal GQA attention with ALiBi. grid = (T, H, B), block = 64 lanes.
// qkv: fp32 [M, 1536] rows = b*T+t, cols = [Q(1024) | K(256) | V(256)].
// Writes bf16 output [M, D] (feeds proj GEMM directly).
// ---------------------------------------------------------------------------
__global__ __launch_bounds__(64)
void attn_kernel(const float* __restrict__ qkv, bf16* __restrict__ o) {
  __shared__ float qs[kHD];
  __shared__ float sc[kT];
  __shared__ float red[64];

  const int tq = blockIdx.x, h = blockIdx.y, b = blockIdx.z;
  const int tid = threadIdx.x;
  const int kvh = h / (kH / kKVH);
  const size_t rowQ = (size_t)(b * kT + tq) * kQKVW;

  qs[tid] = qkv[rowQ + h * kHD + tid];
  __syncthreads();

  const float slope = exp2f(-0.5f * (float)(h + 1));   // ALiBi, H=16 power of 2
  const float scale = 0.125f;                          // 1/sqrt(64)

  float lmax = -3.402823466e38f;
  for (int j = tid; j <= tq; j += 64) {
    const float* kr = qkv + (size_t)(b * kT + j) * kQKVW + kD + kvh * kHD;
    float s = 0.f;
#pragma unroll
    for (int d = 0; d < kHD; ++d) s += qs[d] * kr[d];
    s = s * scale + slope * (float)(j - tq);           // rel = min(j-t,0), j<=t
    sc[j] = s;
    lmax = fmaxf(lmax, s);
  }
  red[tid] = lmax; __syncthreads();
#pragma unroll
  for (int off = 32; off > 0; off >>= 1) {
    if (tid < off) red[tid] = fmaxf(red[tid], red[tid + off]);
    __syncthreads();
  }
  const float mx = red[0];
  __syncthreads();

  float lsum = 0.f;
  for (int j = tid; j <= tq; j += 64) {
    float e = __expf(sc[j] - mx);
    sc[j] = e;
    lsum += e;
  }
  red[tid] = lsum; __syncthreads();
#pragma unroll
  for (int off = 32; off > 0; off >>= 1) {
    if (tid < off) red[tid] += red[tid + off];
    __syncthreads();
  }
  const float rs = 1.0f / red[0];

  // each lane owns output dim d = tid; V reads coalesced across lanes
  const float* vb = qkv + kD + kKVW + (size_t)kvh * kHD + tid;
  float accv = 0.f;
  for (int j = 0; j <= tq; ++j)
    accv += sc[j] * vb[(size_t)(b * kT + j) * kQKVW];
  o[(size_t)(b * kT + tq) * kD + h * kHD + tid] = (bf16)(accv * rs);
}

// ---------------------------------------------------------------------------
// Host-side orchestration
// ---------------------------------------------------------------------------
static void run_gemm(const bf16* A, const bf16* Bm, const float* bias, const float* res,
                     float* outF, bf16* outB, int M, int N, int K,
                     bool transB, bool gelu, bool resid, hipStream_t stream) {
  dim3 grid(N / BN, M / BM), block(256);
  if (transB)
    gemm_wmma_kernel<true, false, false><<<grid, block, 0, stream>>>(A, Bm, bias, res, outF, outB, M, N, K);
  else if (gelu)
    gemm_wmma_kernel<false, true, false><<<grid, block, 0, stream>>>(A, Bm, bias, res, outF, outB, M, N, K);
  else if (resid)
    gemm_wmma_kernel<false, false, true><<<grid, block, 0, stream>>>(A, Bm, bias, res, outF, outB, M, N, K);
  else
    gemm_wmma_kernel<false, false, false><<<grid, block, 0, stream>>>(A, Bm, bias, res, outF, outB, M, N, K);
}

extern "C" void kernel_launch(void* const* d_in, const int* in_sizes, int n_in,
                              void* d_out, int out_size, void* d_ws, size_t ws_size,
                              hipStream_t stream) {
  (void)in_sizes; (void)n_in; (void)out_size; (void)ws_size;

  const float* tok_emb = (const float*)d_in[0];
  const float* ln1_s   = (const float*)d_in[1];
  const float* ln1_b   = (const float*)d_in[2];
  const float* qkv_w   = (const float*)d_in[3];
  const float* qkv_b   = (const float*)d_in[4];
  const float* proj_w  = (const float*)d_in[5];
  const float* proj_b  = (const float*)d_in[6];
  const float* ln2_s   = (const float*)d_in[7];
  const float* ln2_b   = (const float*)d_in[8];
  const float* fc1_w   = (const float*)d_in[9];
  const float* fc1_b   = (const float*)d_in[10];
  const float* fc2_w   = (const float*)d_in[11];
  const float* fc2_b   = (const float*)d_in[12];
  const float* lnf_s   = (const float*)d_in[13];
  const float* lnf_b   = (const float*)d_in[14];
  const int*   idx     = (const int*)d_in[15];
  float* logits = (float*)d_out;

  // workspace carve-up (256B aligned)
  char* p = (char*)d_ws;
  auto take = [&](size_t bytes) {
    char* r = p;
    p += (bytes + 255) & ~(size_t)255;
    return r;
  };
  bf16* tokB   = (bf16*)take((size_t)kV * kD * 2);
  bf16* qkvwB  = (bf16*)take((size_t)kL * kD * kQKVW * 2);
  bf16* projwB = (bf16*)take((size_t)kL * kD * kD * 2);
  bf16* fc1wB  = (bf16*)take((size_t)kL * kD * 4 * kD * 2);
  bf16* fc2wB  = (bf16*)take((size_t)kL * 4 * kD * kD * 2);
  float* x     = (float*)take((size_t)kM * kD * 4);       // residual stream
  bf16* hA     = (bf16*)take((size_t)kM * kD * 2);        // LN outputs (GEMM A)
  bf16* attnB  = (bf16*)take((size_t)kM * kD * 2);        // attention out (bf16)
  bf16* actB   = (bf16*)take((size_t)kM * 4 * kD * 2);    // fc1 out (bf16)
  float* qkvO  = (float*)take((size_t)kM * kQKVW * 4);    // QKV projections

  // one-shot bf16 weight mirrors
  cast_f32_bf16<<<4096, 256, 0, stream>>>(tok_emb, tokB,  (size_t)kV * kD);
  cast_f32_bf16<<<4096, 256, 0, stream>>>(qkv_w,  qkvwB,  (size_t)kL * kD * kQKVW);
  cast_f32_bf16<<<4096, 256, 0, stream>>>(proj_w, projwB, (size_t)kL * kD * kD);
  cast_f32_bf16<<<4096, 256, 0, stream>>>(fc1_w,  fc1wB,  (size_t)kL * kD * 4 * kD);
  cast_f32_bf16<<<4096, 256, 0, stream>>>(fc2_w,  fc2wB,  (size_t)kL * 4 * kD * kD);

  // x = tok_emb[idx]
  embed_kernel<<<kM, 256, 0, stream>>>(tok_emb, idx, x);

  for (int l = 0; l < kL; ++l) {
    // h = LN1(x) -> bf16
    ln_bf16_kernel<<<kM, 256, 0, stream>>>(x, ln1_s + l * kD, ln1_b + l * kD, hA);
    // qkv = h @ qkv_w + qkv_b
    run_gemm(hA, qkvwB + (size_t)l * kD * kQKVW, qkv_b + (size_t)l * kQKVW,
             nullptr, qkvO, nullptr, kM, kQKVW, kD, false, false, false, stream);
    // causal GQA attention + ALiBi -> bf16
    attn_kernel<<<dim3(kT, kH, kB), 64, 0, stream>>>(qkvO, attnB);
    // x += attn @ proj_w + proj_b
    run_gemm(attnB, projwB + (size_t)l * kD * kD, proj_b + (size_t)l * kD,
             x, x, nullptr, kM, kD, kD, false, false, true, stream);
    // h = LN2(x) -> bf16
    ln_bf16_kernel<<<kM, 256, 0, stream>>>(x, ln2_s + l * kD, ln2_b + l * kD, hA);
    // act = gelu(h @ fc1_w + fc1_b) -> bf16
    run_gemm(hA, fc1wB + (size_t)l * kD * 4 * kD, fc1_b + (size_t)l * 4 * kD,
             nullptr, nullptr, actB, kM, 4 * kD, kD, false, true, false, stream);
    // x += act @ fc2_w + fc2_b
    run_gemm(actB, fc2wB + (size_t)l * 4 * kD * kD, fc2_b + (size_t)l * kD,
             x, x, nullptr, kM, kD, 4 * kD, false, false, true, stream);
  }

  // final LN + tied head: logits = LNf(x) @ tok_emb.T  (TRANSB, no bias)
  ln_bf16_kernel<<<kM, 256, 0, stream>>>(x, lnf_s, lnf_b, hA);
  run_gemm(hA, tokB, nullptr, nullptr, logits, nullptr, kM, kV, kD,
           true, false, false, stream);
}
